// NonLocalBlock3D_35905926594683
// MI455X (gfx1250) — matrix operand order repaired and checked by reading, not compiled, not run on hardware
//
#include <hip/hip_runtime.h>
#include <hip/hip_bf16.h>

typedef __bf16 bf16;
typedef __attribute__((ext_vector_type(16))) __bf16 v16bf;
typedef __attribute__((ext_vector_type(8)))  __bf16 v8bf;
typedef __attribute__((ext_vector_type(8)))  float  v8f;

#define WMMA_BF16(a, b, c) \
    __builtin_amdgcn_wmma_f32_16x16x32_bf16(false, (a), false, (b), (short)0, (c), false, false)

// Problem dims (fixed by the reference)
constexpr int BB = 8, C = 256, CI = 128, TT = 8, HH = 28, WW = 28;
constexpr int NN = TT * HH * WW;              // 6272
constexpr int MM = TT * (HH / 2) * (WW / 2);  // 1568

// Workspace layout (bytes, 256-aligned)
constexpr size_t OFF_WCAT  = 0;                          // 384x256 bf16 = 196608
constexpr size_t OFF_WW    = 196608;                     // 256x128 bf16 =  65536
constexpr size_t OFF_THETA = 262144;                     // B*N*CI bf16  = 12845056
constexpr size_t OFF_PFULL = OFF_THETA + 12845056;       // B*N*CI bf16
constexpr size_t OFF_GFULL = OFF_PFULL + 12845056;       // B*N*CI bf16
constexpr size_t OFF_PHI   = OFF_GFULL + 12845056;       // B*M*CI bf16  = 3211264
constexpr size_t OFF_GT    = OFF_PHI   + 3211264;        // B*CI*M bf16  = 3211264
constexpr size_t OFF_Y     = OFF_PFULL;                  // alias (pfull dead after pooling)

// Build a 16x32 bf16 A-fragment for one lane.
// Lane row m = lane&15; element e holds K = 16*half + e (contiguous order,
// matching the B fragments). Two contiguous 16B chunks.
__device__ __forceinline__ v16bf load_a_frag(const bf16* row_kbase, int half) {
    v8bf lo = *(const v8bf*)(row_kbase + 16 * half);
    v8bf hi = *(const v8bf*)(row_kbase + 16 * half + 8);
    return __builtin_shufflevector(lo, hi, 0, 1, 2, 3, 4, 5, 6, 7,
                                           8, 9, 10, 11, 12, 13, 14, 15);
}

__device__ __forceinline__ float rowmax16(float v) {
    for (int m = 1; m < 16; m <<= 1) v = fmaxf(v, __shfl_xor(v, m, 16));
    return v;
}
__device__ __forceinline__ float rowsum16(float v) {
    for (int m = 1; m < 16; m <<= 1) v += __shfl_xor(v, m, 16);
    return v;
}

// ---------------- Kernel 0: weight conversion to bf16 ----------------
// wcat rows: [0,128) = theta_w, [128,256) = phi_w, [256,384) = g_w
__global__ void k_convert(const float* __restrict__ g_w, const float* __restrict__ t_w,
                          const float* __restrict__ p_w, const float* __restrict__ w_w,
                          bf16* __restrict__ wcat, bf16* __restrict__ wwb) {
    int i = blockIdx.x * 256 + threadIdx.x;
    if (i < 384 * 256) {
        int r = i >> 8, c = i & 255;
        float v = (r < 128) ? t_w[r * 256 + c]
                : (r < 256) ? p_w[(r - 128) * 256 + c]
                            : g_w[(r - 256) * 256 + c];
        wcat[i] = (bf16)v;
    } else {
        int j = i - 384 * 256;
        if (j < 256 * 128) wwb[j] = (bf16)w_w[j];
    }
}

// ---------------- Kernel 1: fused projections (theta/phi/g), full res ----------------
// Per block: batch b, 16 output columns (n). GEMM wcat(384x256) @ x(256xN).
// A fragments for a full row-tile are preloaded into distinct registers so the
// compiler can clause the loads and overlap them with the WMMA chain.
__global__ __launch_bounds__(32) void k_proj(const float* __restrict__ x,
                                             const bf16* __restrict__ wcat,
                                             const float* __restrict__ t_b,
                                             const float* __restrict__ p_b,
                                             const float* __restrict__ g_b,
                                             bf16* __restrict__ theta,
                                             bf16* __restrict__ pfull,
                                             bf16* __restrict__ gfull) {
    const int b = blockIdx.y, n0 = blockIdx.x * 16;
    const int lane = threadIdx.x & 31, half = lane >> 4, cm = lane & 15;

    // B-fragments: column n = n0 + cm, element e -> k = ks*32 + 16*half + e
    v16bf bx[8];
    const float* xcol = x + (size_t)b * C * NN + n0 + cm;
    for (int ks = 0; ks < 8; ++ks) {
        v16bf f{};
        for (int e = 0; e < 16; ++e) {
            int k = ks * 32 + 16 * half + e;
            f[e] = (bf16)xcol[(size_t)k * NN];
        }
        bx[ks] = f;
    }

    const size_t orow = (size_t)(b * NN + n0 + cm) * CI;
    for (int ot = 0; ot < 24; ++ot) {
        const bf16* wrow = wcat + (size_t)(ot * 16 + cm) * C;
        v16bf a[8];
#pragma unroll
        for (int ks = 0; ks < 8; ++ks) a[ks] = load_a_frag(wrow + ks * 32, half);

        v8f acc = {};
#pragma unroll
        for (int ks = 0; ks < 8; ++ks) acc = WMMA_BF16(a[ks], bx[ks], acc);

        for (int i = 0; i < 8; ++i) {
            int o = ot * 16 + i + 8 * half;  // out-channel (D layout row)
            float bias = (o < 128) ? t_b[o] : (o < 256) ? p_b[o - 128] : g_b[o - 256];
            float v = acc[i] + bias;
            if (o < 128)      theta[orow + o]         = (bf16)v;
            else if (o < 256) pfull[orow + (o - 128)] = (bf16)v;
            else              gfull[orow + (o - 256)] = (bf16)v;
        }
    }
}

// ---------------- Kernel 2: 2x2 max-pool; g stored transposed (ci, m) ----------------
__global__ void k_pool(const bf16* __restrict__ pfull, const bf16* __restrict__ gfull,
                       bf16* __restrict__ phi, bf16* __restrict__ gt) {
    int idx = blockIdx.x * 256 + threadIdx.x;
    if (idx >= BB * MM * CI) return;
    int ci = idx & (CI - 1);
    int m  = (idx >> 7) % MM;
    int b  = idx / (MM * CI);
    int t = m / 196, r = m % 196, h2 = r / 14, w2 = r % 14;
    int n00 = t * 784 + (2 * h2) * 28 + 2 * w2;
    size_t base = (size_t)b * NN * CI + (size_t)ci;
    float p00 = (float)pfull[base + (size_t)(n00)      * CI];
    float p01 = (float)pfull[base + (size_t)(n00 + 1)  * CI];
    float p10 = (float)pfull[base + (size_t)(n00 + 28) * CI];
    float p11 = (float)pfull[base + (size_t)(n00 + 29) * CI];
    float g00 = (float)gfull[base + (size_t)(n00)      * CI];
    float g01 = (float)gfull[base + (size_t)(n00 + 1)  * CI];
    float g10 = (float)gfull[base + (size_t)(n00 + 28) * CI];
    float g11 = (float)gfull[base + (size_t)(n00 + 29) * CI];
    phi[((size_t)(b * MM + m)) * CI + ci] = (bf16)fmaxf(fmaxf(p00, p01), fmaxf(p10, p11));
    gt[((size_t)(b * CI + ci)) * MM + m]  = (bf16)fmaxf(fmaxf(g00, g01), fmaxf(g10, g11));
}

// ---------------- Kernel 3: fused attention (flash-style online softmax) ----------------
// One wave per (batch, 16 N-rows). S = theta@phiT in tiles of 16x32; softmax
// accumulated online; O += P@g with P routed D-layout -> A-layout through LDS.
// gt B-fragments are loaded BEFORE the P/LDS round trip so global latency
// overlaps the softmax VALU work and the ds wait.
__global__ __launch_bounds__(32) void k_attn(const bf16* __restrict__ theta,
                                             const bf16* __restrict__ phi,
                                             const bf16* __restrict__ gt,
                                             bf16* __restrict__ y) {
    const int b = blockIdx.y, n0 = blockIdx.x * 16;
    const int lane = threadIdx.x & 31, half = lane >> 4, cm = lane & 15;
    __shared__ alignas(32) bf16 lp[16 * 32];

    // Resident theta A-fragments (16 rows x 128 K)
    v16bf at[4];
    const bf16* trow = theta + (size_t)(b * NN + n0 + cm) * CI;
#pragma unroll
    for (int ks = 0; ks < 4; ++ks) at[ks] = load_a_frag(trow + ks * 32, half);

    float mr[8], lr[8];
    v8f O[8];
    for (int i = 0; i < 8; ++i) { mr[i] = -INFINITY; lr[i] = 0.0f; }
    for (int c = 0; c < 8; ++c) O[c] = {};

    const bf16* phirow0 = phi + (size_t)(b * MM + cm) * CI + 16 * half;
    const bf16* gtbase  = gt + (size_t)(b * CI + cm) * MM + 16 * half;

    for (int mc = 0; mc < MM / 32; ++mc) {
        const int m0 = mc * 32;

        // Prefetch next chunk's phi/gt rows into cache (global_prefetch_b8).
        if (mc + 1 < MM / 32) {
            __builtin_prefetch(phirow0 + (size_t)(m0 + 32) * CI, 0, 3);
            __builtin_prefetch(gtbase + m0 + 32, 0, 3);
        }

        v8f s0 = {}, s1 = {};
        // phi rows m0+cm and m0+16+cm; element e -> ci = ks*32 + 16*half + e (contiguous)
        const bf16* pr0 = phirow0 + (size_t)m0 * CI;
        const bf16* pr1 = pr0 + (size_t)16 * CI;
#pragma unroll
        for (int ks = 0; ks < 4; ++ks) {
            v16bf b0 = *(const v16bf*)(pr0 + ks * 32);
            v16bf b1 = *(const v16bf*)(pr1 + ks * 32);
            s0 = WMMA_BF16(at[ks], b0, s0);
            s1 = WMMA_BF16(at[ks], b1, s1);
        }

        // Issue all gt B-fragment loads now; they are independent of P and
        // overlap the softmax + LDS round trip below.
        v16bf gf[8];
#pragma unroll
        for (int c = 0; c < 8; ++c) {
            gf[c] = *(const v16bf*)(gtbase + (size_t)(16 * c) * MM + m0);
        }

        // Online softmax over this 32-column chunk (rows live in (vgpr, lane-half))
        float pa0[8], pa1[8];
        for (int i = 0; i < 8; ++i) {
            float mx = rowmax16(fmaxf(s0[i], s1[i]));
            float mn = fmaxf(mr[i], mx);
            float al = __expf(mr[i] - mn);
            mr[i] = mn;
            float e0 = __expf(s0[i] - mn);
            float e1 = __expf(s1[i] - mn);
            lr[i] = lr[i] * al + rowsum16(e0 + e1);
            pa0[i] = e0; pa1[i] = e1;
            for (int c = 0; c < 8; ++c) O[c][i] *= al;
        }

        // P (16x32, D layout) -> LDS -> A-fragment layout
        for (int i = 0; i < 8; ++i) {
            lp[(i + 8 * half) * 32 + cm]      = (bf16)pa0[i];
            lp[(i + 8 * half) * 32 + cm + 16] = (bf16)pa1[i];
        }
        asm volatile("s_wait_dscnt 0" ::: "memory");
        v16bf pfrag = load_a_frag(lp + cm * 32, half);

        // O(16x128) += P(16x32) @ g(32x128)
#pragma unroll
        for (int c = 0; c < 8; ++c) O[c] = WMMA_BF16(pfrag, gf[c], O[c]);
    }

    // Normalize and store y as (n, ci) bf16
    for (int i = 0; i < 8; ++i) {
        float inv = 1.0f / lr[i];
        size_t yrow = (size_t)(b * NN + n0 + i + 8 * half) * CI;
        for (int c = 0; c < 8; ++c) {
            y[yrow + 16 * c + cm] = (bf16)(O[c][i] * inv);
        }
    }
}

// ---------------- Kernel 4: output projection + bias + residual ----------------
// Two row-tiles per iteration with preloaded A fragments for load/WMMA overlap.
__global__ __launch_bounds__(32) void k_wout(const bf16* __restrict__ y,
                                             const bf16* __restrict__ wwb,
                                             const float* __restrict__ w_b,
                                             const float* __restrict__ x,
                                             float* __restrict__ out) {
    const int b = blockIdx.y, n0 = blockIdx.x * 16;
    const int lane = threadIdx.x & 31, half = lane >> 4, cm = lane & 15;

    v16bf by[4];
    const bf16* yrow = y + (size_t)(b * NN + n0 + cm) * CI + 16 * half;
#pragma unroll
    for (int ks = 0; ks < 4; ++ks) by[ks] = *(const v16bf*)(yrow + ks * 32);

    for (int ct = 0; ct < 16; ct += 2) {
        const bf16* w0 = wwb + (size_t)(ct * 16 + cm) * CI;
        const bf16* w1 = w0 + (size_t)16 * CI;
        v16bf a0[4], a1[4];
#pragma unroll
        for (int ks = 0; ks < 4; ++ks) a0[ks] = load_a_frag(w0 + ks * 32, half);
#pragma unroll
        for (int ks = 0; ks < 4; ++ks) a1[ks] = load_a_frag(w1 + ks * 32, half);

        v8f acc0 = {}, acc1 = {};
#pragma unroll
        for (int ks = 0; ks < 4; ++ks) acc0 = WMMA_BF16(a0[ks], by[ks], acc0);
#pragma unroll
        for (int ks = 0; ks < 4; ++ks) acc1 = WMMA_BF16(a1[ks], by[ks], acc1);

        for (int i = 0; i < 8; ++i) {
            int c0 = ct * 16 + i + 8 * half;
            int c1 = c0 + 16;
            size_t off0 = (size_t)(b * C + c0) * NN + n0 + cm;
            size_t off1 = (size_t)(b * C + c1) * NN + n0 + cm;
            out[off0] = acc0[i] + w_b[c0] + x[off0];
            out[off1] = acc1[i] + w_b[c1] + x[off1];
        }
    }
}

extern "C" void kernel_launch(void* const* d_in, const int* in_sizes, int n_in,
                              void* d_out, int out_size, void* d_ws, size_t ws_size,
                              hipStream_t stream) {
    const float* x   = (const float*)d_in[0];
    const float* g_w = (const float*)d_in[1];
    const float* g_b = (const float*)d_in[2];
    const float* t_w = (const float*)d_in[3];
    const float* t_b = (const float*)d_in[4];
    const float* p_w = (const float*)d_in[5];
    const float* p_b = (const float*)d_in[6];
    const float* W_w = (const float*)d_in[7];
    const float* W_b = (const float*)d_in[8];
    float* out = (float*)d_out;

    char* ws = (char*)d_ws;
    bf16* wcat  = (bf16*)(ws + OFF_WCAT);
    bf16* wwb   = (bf16*)(ws + OFF_WW);
    bf16* theta = (bf16*)(ws + OFF_THETA);
    bf16* pfull = (bf16*)(ws + OFF_PFULL);
    bf16* gfull = (bf16*)(ws + OFF_GFULL);
    bf16* phi   = (bf16*)(ws + OFF_PHI);
    bf16* gt    = (bf16*)(ws + OFF_GT);
    bf16* y     = (bf16*)(ws + OFF_Y);   // aliases pfull (dead after pooling)

    // 1) weights -> bf16
    k_convert<<<dim3((384 * 256 + 256 * 128) / 256), dim3(256), 0, stream>>>(
        g_w, t_w, p_w, W_w, wcat, wwb);

    // 2) fused theta/phi/g projections (WMMA)
    k_proj<<<dim3(NN / 16, BB), dim3(32), 0, stream>>>(
        x, wcat, t_b, p_b, g_b, theta, pfull, gfull);

    // 3) 2x2 spatial max-pool (g transposed)
    k_pool<<<dim3((BB * MM * CI) / 256), dim3(256), 0, stream>>>(pfull, gfull, phi, gt);

    // 4) fused attention with online softmax (WMMA)
    k_attn<<<dim3(NN / 16, BB), dim3(32), 0, stream>>>(theta, phi, gt, y);

    // 5) output projection + bias + residual (WMMA)
    k_wout<<<dim3(NN / 16, BB), dim3(32), 0, stream>>>(y, wwb, W_b, x, out);
}
